// Model_75402445848602
// MI455X (gfx1250) — matrix operand order repaired and checked by reading, not compiled, not run on hardware
//
#include <hip/hip_runtime.h>
#include <hip/hip_bf16.h>
#include <math.h>

typedef __attribute__((ext_vector_type(16))) _Float16 v16h;
typedef __attribute__((ext_vector_type(8)))  float    v8f;

#define HID   32
#define HEADS 8
#define HC    256   // HEADS*HID

// ---------- order-preserving float <-> uint mapping for atomic max ----------
__device__ __forceinline__ unsigned f2ord(float f) {
  unsigned u = __float_as_uint(f);
  return (u & 0x80000000u) ? ~u : (u | 0x80000000u);
}
__device__ __forceinline__ float ord2f(unsigned u) {
  unsigned b = (u & 0x80000000u) ? (u ^ 0x80000000u) : ~u;
  return __uint_as_float(b);
}
#define ORD_NEG_INF 0x007fffffu   // f2ord(-inf)

// ---------------------------------------------------------------------------
// Pack B[K,Nreal] (f32, row-major) into WMMA B-fragment order, f16:
//   Bp[((kblk*tilesN)+tn)*32 + lane][j] = B[kblk*32 + (j>>3)*16 + (lane>>4)*8 + (j&7), tn*16+(lane&15)]
// cols >= Nreal are zero-padded (handles lin5's N=8 in a 16-wide tile).
// ---------------------------------------------------------------------------
__global__ void pack_b_kernel(const float* __restrict__ B, v16h* __restrict__ Bp,
                              int K, int Nreal, int tilesN) {
  int t = blockIdx.x * blockDim.x + threadIdx.x;
  int total = (K / 32) * tilesN * 32;
  if (t >= total) return;
  int lane = t & 31;
  int tmp  = t >> 5;
  int tn   = tmp % tilesN;
  int kblk = tmp / tilesN;
  int half = lane >> 4, r = lane & 15;
  int col  = tn * 16 + r;
  v16h frag;
#pragma unroll
  for (int j = 0; j < 16; ++j) {
    int krow = kblk * 32 + (j >> 3) * 16 + half * 8 + (j & 7);
    float v  = (col < Nreal) ? B[(size_t)krow * Nreal + col] : 0.0f;
    frag[j]  = (_Float16)v;
  }
  Bp[t] = frag;
}

__global__ void pad_bias_kernel(const float* __restrict__ b, float* __restrict__ bp,
                                int Nreal, int Npad) {
  int t = threadIdx.x;
  if (t < Npad) bp[t] = (t < Nreal) ? b[t] : 0.0f;
}

// ---------------------------------------------------------------------------
// WMMA GEMM: C[M,NCOLS] = A[M,K] @ B (pre-packed f16 fragments) (+bias).
// M % 16 == 0. Compile-time K/NCOLS/TILESN -> fully unrolled K loop,
// immediate-offset stores. 8 waves/block, wave = one 16x16 tile.
// ---------------------------------------------------------------------------
template <int K, int NCOLS, int TILESN>
__global__ __launch_bounds__(256)
void wmma_gemm_kernel(const float* __restrict__ A, const v16h* __restrict__ Bp,
                      const float* __restrict__ bias, float* __restrict__ C,
                      int M) {
  const int waveId = blockIdx.x * 8 + (threadIdx.x >> 5);
  const int tileM  = waveId / TILESN;
  const int tileN  = waveId - tileM * TILESN;
  if (tileM * 16 >= M) return;            // wave-uniform exit, EXEC stays full
  const int lane = threadIdx.x & 31;
  const int half = lane >> 4;             // 0: lanes 0-15, 1: lanes 16-31
  const int r    = lane & 15;

  const float* __restrict__ arow = A + (size_t)(tileM * 16 + r) * K;
  const v16h*  __restrict__ bp   = Bp + (size_t)tileN * 32 + lane;

  v8f acc = {};
#pragma unroll
  for (int kb = 0; kb < K; kb += 32) {
    // A fragment (ISA 16-bit 16x32): two contiguous 8-float spans per lane
    const float4 a0 = *(const float4*)(arow + kb + half * 8);
    const float4 a1 = *(const float4*)(arow + kb + half * 8 + 4);
    const float4 a2 = *(const float4*)(arow + kb + 16 + half * 8);
    const float4 a3 = *(const float4*)(arow + kb + 16 + half * 8 + 4);
    v16h af;
    af[0]  = (_Float16)a0.x; af[1]  = (_Float16)a0.y;
    af[2]  = (_Float16)a0.z; af[3]  = (_Float16)a0.w;
    af[4]  = (_Float16)a1.x; af[5]  = (_Float16)a1.y;
    af[6]  = (_Float16)a1.z; af[7]  = (_Float16)a1.w;
    af[8]  = (_Float16)a2.x; af[9]  = (_Float16)a2.y;
    af[10] = (_Float16)a2.z; af[11] = (_Float16)a2.w;
    af[12] = (_Float16)a3.x; af[13] = (_Float16)a3.y;
    af[14] = (_Float16)a3.z; af[15] = (_Float16)a3.w;

    // B fragment: one contiguous 32-byte vector load in fragment order
    const v16h bf = bp[(size_t)(kb >> 5) * (TILESN * 32)];

    acc = __builtin_amdgcn_wmma_f32_16x16x32_f16(
        /*neg_a=*/false, af, /*neg_b=*/false, bf,
        /*c_mod=*/(short)0, acc, /*reuse_a=*/false, /*reuse_b=*/false);
  }

  const float bv = bias ? bias[tileN * 16 + r] : 0.0f;
  float* __restrict__ crow =
      C + (size_t)(tileM * 16 + half * 8) * NCOLS + tileN * 16 + r;
#pragma unroll
  for (int v = 0; v < 8; ++v)             // D: VGPR v -> row half*8 + v
    crow[(size_t)v * NCOLS] = acc[v] + bv;
}

// ------------------------------- graph kernels ------------------------------
__global__ void gather_kernel(const float* __restrict__ emb,
                              const int* __restrict__ ids,
                              float* __restrict__ x, int N) {
  int t = blockIdx.x * blockDim.x + threadIdx.x;
  if (t >= N * HID) return;
  int n = t / HID, c = t % HID;
  x[t] = emb[(size_t)ids[n] * HID + c];
}

// al_src[n,h] = sum_c h[n,h,c]*a_src[h,c] ; same for al_dst
__global__ void node_logits_kernel(const float* __restrict__ h,
                                   const float* __restrict__ a_src,
                                   const float* __restrict__ a_dst,
                                   float* __restrict__ al_src,
                                   float* __restrict__ al_dst, int N) {
  int t = blockIdx.x * blockDim.x + threadIdx.x;
  if (t >= N * HEADS) return;
  int n = t / HEADS, hh = t % HEADS;
  const float4* hp = (const float4*)(h + (size_t)n * HC + hh * HID);
  const float4* as = (const float4*)(a_src + hh * HID);
  const float4* ad = (const float4*)(a_dst + hh * HID);
  float s = 0.f, d = 0.f;
#pragma unroll
  for (int q = 0; q < HID / 4; ++q) {
    float4 v = hp[q], a = as[q], b = ad[q];
    s += v.x * a.x + v.y * a.y + v.z * a.z + v.w * a.w;
    d += v.x * b.x + v.y * b.y + v.z * b.z + v.w * b.w;
  }
  al_src[t] = s;
  al_dst[t] = d;
}

// M[k,h] = sum_c We[k, h*HID+c] * a_e[h,c]  -- collapses the E x 256 he matrix
__global__ void edge_M_kernel(const float* __restrict__ We,
                              const float* __restrict__ a_e,
                              float* __restrict__ M) {
  int t = threadIdx.x;
  if (t >= HEADS * HEADS) return;
  int k = t / HEADS, hh = t % HEADS;
  float s = 0.f;
#pragma unroll
  for (int c = 0; c < HID; ++c) s += We[k * HC + hh * HID + c] * a_e[hh * HID + c];
  M[t] = s;
}

__global__ void init_kernel(float* __restrict__ agg, float* __restrict__ denom,
                            unsigned* __restrict__ amax, int N) {
  int t = blockIdx.x * blockDim.x + threadIdx.x;
  if (t >= N * HC) return;
  agg[t] = 0.f;
  if (t < N * HEADS) { denom[t] = 0.f; amax[t] = ORD_NEG_INF; }
}

// one thread per EDGE: alpha[e,h] = leaky(al_src[s,h]+al_dst[d,h]+eattr[e]@M[:,h])
__global__ void alpha_max_kernel(const int* __restrict__ src,
                                 const int* __restrict__ dst,
                                 const float* __restrict__ eattr,
                                 const float* __restrict__ Mm,
                                 const float* __restrict__ al_src,
                                 const float* __restrict__ al_dst,
                                 float* __restrict__ alpha,
                                 unsigned* __restrict__ amax, int E) {
  int e = blockIdx.x * blockDim.x + threadIdx.x;
  if (e >= E) return;
  const int s = src[e], d = dst[e];
  float ea[HEADS];
  const float4* ep = (const float4*)(eattr + (size_t)e * HEADS);
  float4 e0 = ep[0], e1 = ep[1];
  ea[0] = e0.x; ea[1] = e0.y; ea[2] = e0.z; ea[3] = e0.w;
  ea[4] = e1.x; ea[5] = e1.y; ea[6] = e1.z; ea[7] = e1.w;
#pragma unroll
  for (int hh = 0; hh < HEADS; ++hh) {
    float ale = 0.f;
#pragma unroll
    for (int k = 0; k < HEADS; ++k) ale += ea[k] * Mm[k * HEADS + hh];
    float a = al_src[(size_t)s * HEADS + hh] + al_dst[(size_t)d * HEADS + hh] + ale;
    a = (a > 0.f) ? a : 0.2f * a;   // leaky_relu 0.2
    alpha[(size_t)e * HEADS + hh] = a;
    atomicMax(&amax[(size_t)d * HEADS + hh], f2ord(a));
  }
}

// one thread per EDGE: ex = exp(alpha - amax[dst]); denom[dst] += ex
__global__ void ex_denom_kernel(const int* __restrict__ dst,
                                float* __restrict__ alpha,
                                const unsigned* __restrict__ amax,
                                float* __restrict__ denom, int E) {
  int e = blockIdx.x * blockDim.x + threadIdx.x;
  if (e >= E) return;
  const int d = dst[e];
#pragma unroll
  for (int hh = 0; hh < HEADS; ++hh) {
    float m  = ord2f(amax[(size_t)d * HEADS + hh]);
    float ex = __expf(alpha[(size_t)e * HEADS + hh] - m);
    alpha[(size_t)e * HEADS + hh] = ex;
    atomicAdd(&denom[(size_t)d * HEADS + hh], ex);
  }
}

// 4 edges per 256-thread block; 64 lanes per edge; float4 gather of h[src]
__global__ __launch_bounds__(256)
void aggregate_kernel(const int* __restrict__ src, const int* __restrict__ dst,
                      const float* __restrict__ h, const float* __restrict__ ex,
                      const float* __restrict__ denom, float* __restrict__ agg,
                      int E) {
  int e = blockIdx.x * 4 + (threadIdx.x >> 6);
  if (e >= E) return;
  int t  = threadIdx.x & 63;
  int hh = t >> 3;                 // head 0..7
  int c0 = (t & 7) << 2;           // channel base 0,4,...,28
  int s = src[e], d = dst[e];
  float w = ex[(size_t)e * HEADS + hh] / (denom[(size_t)d * HEADS + hh] + 1e-16f);
  const float4 hv = *(const float4*)(h + (size_t)s * HC + hh * HID + c0);
  float* ap = agg + (size_t)d * HC + hh * HID + c0;
  atomicAdd(ap + 0, hv.x * w);
  atomicAdd(ap + 1, hv.y * w);
  atomicAdd(ap + 2, hv.z * w);
  atomicAdd(ap + 3, hv.w * w);
}

__global__ void bias_relu_kernel(float* __restrict__ agg,
                                 const float* __restrict__ b, int N) {
  int t = blockIdx.x * blockDim.x + threadIdx.x;
  if (t >= N * HC) return;
  float v = agg[t] + b[t & (HC - 1)];
  agg[t] = v > 0.f ? v : 0.f;
}

// fu = [xf[u], elattr[l]] @ clfW + clfb ; out = sigmoid(dot(fu, xf[m]))
__global__ void classifier_kernel(const int* __restrict__ eli,
                                  const float* __restrict__ elattr,
                                  const float* __restrict__ xf, int xf_stride,
                                  const float* __restrict__ clfW,
                                  const float* __restrict__ clfb,
                                  float* __restrict__ out, int L) {
  int l = blockIdx.x * blockDim.x + threadIdx.x;
  if (l >= L) return;
  int u = eli[l];
  int m = eli[L + l];
  float f[16];
#pragma unroll
  for (int i = 0; i < 8; ++i) f[i] = xf[(size_t)u * xf_stride + i];
#pragma unroll
  for (int i = 0; i < 8; ++i) f[8 + i] = elattr[(size_t)l * 8 + i];
  float z = 0.f;
#pragma unroll
  for (int j = 0; j < 8; ++j) {
    float fu = clfb[j];
#pragma unroll
    for (int i = 0; i < 16; ++i) fu += f[i] * clfW[i * 8 + j];
    z += fu * xf[(size_t)m * xf_stride + j];
  }
  out[l] = 1.f / (1.f + __expf(-z));
}

// ---------------------------------------------------------------------------
extern "C" void kernel_launch(void* const* d_in, const int* in_sizes, int n_in,
                              void* d_out, int out_size, void* d_ws, size_t ws_size,
                              hipStream_t stream) {
  const int*   node_ids = (const int*)  d_in[0];
  const int*   eidx     = (const int*)  d_in[1];
  const int*   eli      = (const int*)  d_in[2];
  // d_in[3] = topic (unused)
  const float* eattr    = (const float*)d_in[4];
  const float* elattr   = (const float*)d_in[5];
  const float* emb      = (const float*)d_in[6];

  const float* Wl[2]    = {(const float*)d_in[7],  (const float*)d_in[15]};
  const float* a_src[2] = {(const float*)d_in[8],  (const float*)d_in[16]};
  const float* a_dst[2] = {(const float*)d_in[9],  (const float*)d_in[17]};
  const float* Wel[2]   = {(const float*)d_in[10], (const float*)d_in[18]};
  const float* a_el[2]  = {(const float*)d_in[11], (const float*)d_in[19]};
  const float* bl[2]    = {(const float*)d_in[12], (const float*)d_in[20]};
  const float* lin1W    = (const float*)d_in[13];
  const float* lin1b    = (const float*)d_in[14];
  const float* lin5W    = (const float*)d_in[21];
  const float* lin5b    = (const float*)d_in[22];
  const float* clfW     = (const float*)d_in[23];
  const float* clfb     = (const float*)d_in[24];

  const int N = in_sizes[0];
  const int E = in_sizes[1] / 2;
  const int L = in_sizes[2] / 2;
  const int* esrc = eidx;
  const int* edst = eidx + E;

  // ---- workspace carve-up (floats; all offsets 32B-aligned) ----
  float* ws = (float*)d_ws;
  size_t off = 0;
  float*    x0     = ws + off; off += (size_t)N * HID;    // layer input [N,32]
  float*    hbuf   = ws + off; off += (size_t)N * HC;     // h = x@W [N,256]
  float*    agg    = ws + off; off += (size_t)N * HC;     // aggregated [N,256]
  float*    al_src = ws + off; off += (size_t)N * HEADS;
  float*    al_dst = ws + off; off += (size_t)N * HEADS;
  float*    denom  = ws + off; off += (size_t)N * HEADS;
  unsigned* amax   = (unsigned*)(ws + off); off += (size_t)N * HEADS;
  float*    alpha  = ws + off; off += (size_t)E * HEADS;  // alpha, then ex
  float*    Mm     = ws + off; off += 64;
  float*    xf     = ws + off; off += (size_t)N * 16;     // padded [N,16]
  v16h*     Wpack  = (v16h*)(ws + off); off += 512 * 8;   // (32/32)*16*32 frags
  v16h*     L1pack = (v16h*)(ws + off); off += 512 * 8;   // (256/32)*2*32 frags
  v16h*     L5pack = (v16h*)(ws + off); off += 256 * 8;   // (256/32)*1*32 frags
  float*    b5pad  = ws + off; off += 16;
  (void)ws_size; (void)n_in;

  const int TB = 256;
  auto cdiv = [](long long a, long long b) { return (int)((a + b - 1) / b); };
  auto gemm_blocks = [&](int tilesN) {
    return cdiv((long long)(N / 16 + (N % 16 ? 1 : 0)) * tilesN, 8);
  };

  // pre-pack lin weights (per call; deterministic)
  pack_b_kernel<<<cdiv((HC / 32) * 2 * 32, TB), TB, 0, stream>>>(lin1W, L1pack, HC, HID, 2);
  pack_b_kernel<<<cdiv((HC / 32) * 1 * 32, TB), TB, 0, stream>>>(lin5W, L5pack, HC, HEADS, 1);
  pad_bias_kernel<<<1, 16, 0, stream>>>(lin5b, b5pad, HEADS, 16);

  // gather: x0 = emb[node_ids]
  gather_kernel<<<cdiv((long long)N * HID, TB), TB, 0, stream>>>(emb, node_ids, x0, N);

  for (int layer = 0; layer < 2; ++layer) {
    // pack W (32x256) into fragment order, then h = x0 @ W -- WMMA
    pack_b_kernel<<<cdiv((HID / 32) * 16 * 32, TB), TB, 0, stream>>>(
        Wl[layer], Wpack, HID, HC, 16);
    wmma_gemm_kernel<HID, HC, 16><<<gemm_blocks(16), TB, 0, stream>>>(
        x0, Wpack, nullptr, hbuf, N);

    node_logits_kernel<<<cdiv((long long)N * HEADS, TB), TB, 0, stream>>>(
        hbuf, a_src[layer], a_dst[layer], al_src, al_dst, N);
    edge_M_kernel<<<1, 64, 0, stream>>>(Wel[layer], a_el[layer], Mm);
    init_kernel<<<cdiv((long long)N * HC, TB), TB, 0, stream>>>(agg, denom, amax, N);
    alpha_max_kernel<<<cdiv(E, TB), TB, 0, stream>>>(
        esrc, edst, eattr, Mm, al_src, al_dst, alpha, amax, E);
    ex_denom_kernel<<<cdiv(E, TB), TB, 0, stream>>>(edst, alpha, amax, denom, E);
    aggregate_kernel<<<cdiv(E, 4), TB, 0, stream>>>(
        esrc, edst, hbuf, alpha, denom, agg, E);
    bias_relu_kernel<<<cdiv((long long)N * HC, TB), TB, 0, stream>>>(agg, bl[layer], N);

    if (layer == 0) {
      // x0 = agg @ lin1_W + lin1_b  (K=256, N=32) -- WMMA, 8 unrolled MMAs
      wmma_gemm_kernel<HC, HID, 2><<<gemm_blocks(2), TB, 0, stream>>>(
          agg, L1pack, lin1b, x0, N);
    } else {
      // xf = agg @ lin5_W(padded to 16) + lin5_b  (K=256, N=16) -- WMMA
      wmma_gemm_kernel<HC, 16, 1><<<gemm_blocks(1), TB, 0, stream>>>(
          agg, L5pack, b5pad, xf, N);
    }
  }

  classifier_kernel<<<cdiv(L, TB), TB, 0, stream>>>(
      eli, elattr, xf, /*xf_stride=*/16, clfW, clfb, (float*)d_out, L);
}
